// TransformerBlock_2224793059799
// MI455X (gfx1250) — matrix operand order repaired and checked by reading, not compiled, not run on hardware
//
#include <hip/hip_runtime.h>
#include <hip/hip_bf16.h>
#include <cstdint>

// ---------------- problem dims (fixed by reference) ----------------
constexpr int Bb   = 4;
constexpr int Tt   = 2048;
constexpr int Cc   = 1024;
constexpr int Hh   = 16;
constexpr int HDd  = 64;
constexpr int DFFd = 4096;
constexpr int Mrows = Bb * Tt;      // 8192
constexpr int C3    = 3 * Cc;       // 3072

// ---------------- workspace layout (bytes) ----------------
constexpr size_t OFF_WQKVT = 0;                                        // 3072*1024*2 = 6 MiB
constexpr size_t OFF_WOT   = OFF_WQKVT + (size_t)C3 * Cc * 2;          // + 2 MiB
constexpr size_t OFF_WFCT  = OFF_WOT   + (size_t)Cc * Cc * 2;          // + 8 MiB
constexpr size_t OFF_WPRJT = OFF_WFCT  + (size_t)DFFd * Cc * 2;        // + 8 MiB
constexpr size_t OFF_H     = OFF_WPRJT + (size_t)Cc * DFFd * 2;        // h / h2 bf16 16 MiB
constexpr size_t OFF_QKV   = OFF_H     + (size_t)Mrows * Cc * 2;       // qkv bf16 48 MiB (reused as g: 64 MiB)
constexpr size_t OFF_ATTN  = OFF_QKV   + (size_t)Mrows * DFFd * 2;     // attn bf16 16 MiB
constexpr size_t OFF_X1    = OFF_ATTN  + (size_t)Mrows * Cc * 2;       // x1 fp32 32 MiB

// ---------------- types ----------------
typedef __attribute__((ext_vector_type(16))) __bf16 bf16x16;
typedef __attribute__((ext_vector_type(8)))  float  f32x8;
typedef __attribute__((ext_vector_type(4)))  int    i32x4;

union Frag { bf16x16 v; uint4 u[2]; };

// CDNA5 async global->LDS path (ASYNCcnt), guarded so either toolchain compiles.
#if defined(__gfx1250__) && __has_builtin(__builtin_amdgcn_global_load_async_to_lds_b128) && __has_builtin(__builtin_amdgcn_s_wait_asynccnt)
#define USE_ASYNC_LDS 1
typedef __attribute__((address_space(1))) i32x4 as1_i32x4;
typedef __attribute__((address_space(3))) i32x4 as3_i32x4;
#else
#define USE_ASYNC_LDS 0
#endif

__device__ __forceinline__ f32x8 zero8() {
  f32x8 z;
#pragma unroll
  for (int i = 0; i < 8; ++i) z[i] = 0.0f;
  return z;
}

__device__ __forceinline__ uint16_t f2bf(float f) {
  union { float f; uint32_t u; } v; v.f = f;
  uint32_t r = v.u + 0x7FFFu + ((v.u >> 16) & 1u);   // round-to-nearest-even
  return (uint16_t)(r >> 16);
}

__device__ __forceinline__ float gelu_new(float x) {
  const float c = 0.7978845608028654f;               // sqrt(2/pi)
  float t = tanhf(c * (x + 0.044715f * x * x * x));
  return 0.5f * x * (1.0f + t);
}

// =====================================================================
// 1) weight transpose + fp32 -> bf16:  W[K][N] -> Wt[N][K]
// =====================================================================
__global__ __launch_bounds__(256) void transpose_cvt_kernel(
    const float* __restrict__ W, uint16_t* __restrict__ Wt, int K, int N) {
  __shared__ float tile[32][33];
  const int tx = threadIdx.x & 31;
  const int ty = threadIdx.x >> 5;      // 0..7
  const int n0 = blockIdx.x * 32;
  const int k0 = blockIdx.y * 32;
#pragma unroll
  for (int i = 0; i < 4; ++i)
    tile[ty + i * 8][tx] = W[(size_t)(k0 + ty + i * 8) * N + (n0 + tx)];
  __syncthreads();
#pragma unroll
  for (int i = 0; i < 4; ++i)
    Wt[(size_t)(n0 + ty + i * 8) * K + (k0 + tx)] = f2bf(tile[tx][ty + i * 8]);
}

// =====================================================================
// 2) layernorm (fp32 in) -> bf16 out, one 256-thread block per row, C=1024
// =====================================================================
__global__ __launch_bounds__(256) void layernorm_kernel(
    const float* __restrict__ x, const float* __restrict__ g,
    const float* __restrict__ bta, uint16_t* __restrict__ out) {
  const int row = blockIdx.x;
  const float* xr = x + (size_t)row * Cc;
  float v[4], s = 0.f, ss = 0.f;
#pragma unroll
  for (int i = 0; i < 4; ++i) {
    v[i] = xr[threadIdx.x + i * 256];
    s  += v[i];
    ss += v[i] * v[i];
  }
#pragma unroll
  for (int o = 16; o; o >>= 1) {
    s  += __shfl_xor(s,  o, 32);
    ss += __shfl_xor(ss, o, 32);
  }
  __shared__ float ps[8], pq[8];
  const int wv = threadIdx.x >> 5, ln = threadIdx.x & 31;
  if (ln == 0) { ps[wv] = s; pq[wv] = ss; }
  __syncthreads();
  s = 0.f; ss = 0.f;
#pragma unroll
  for (int i = 0; i < 8; ++i) { s += ps[i]; ss += pq[i]; }
  const float mu   = s * (1.0f / Cc);
  const float var  = ss * (1.0f / Cc) - mu * mu;
  const float rstd = rsqrtf(var + 1e-5f);
#pragma unroll
  for (int i = 0; i < 4; ++i) {
    const int c = threadIdx.x + i * 256;
    out[(size_t)row * Cc + c] = f2bf((v[i] - mu) * rstd * g[c] + bta[c]);
  }
}

// =====================================================================
// 3) bf16 WMMA GEMM: Out[M][N] = epilogue(A[M][K] @ Bt[N][K]^T + bias)
//    block = 256 threads (8 waves), tile 128x128, K-step 32
//    double-buffered LDS; async global->LDS DMA when available (ASYNCcnt)
//    wave(w): rows (w>>1)*32, cols (w&1)*64 -> 2x4 sub-tiles of 16x16
// =====================================================================
enum { EPI_BIAS_BF16 = 0, EPI_BIAS_RES_F32 = 1, EPI_BIAS_GELU_BF16 = 2 };

template <int EPI>
__global__ __launch_bounds__(256) void gemm_bf16_kernel(
    const uint16_t* __restrict__ A, const uint16_t* __restrict__ Bt,
    const float* __restrict__ bias, const float* __restrict__ resid,
    void* __restrict__ Out, int M, int N, int K) {
  __shared__ uint16_t lsA[2][128][40];
  __shared__ uint16_t lsB[2][128][40];
  const int tid  = threadIdx.x;
  const int wv   = tid >> 5;
  const int lane = tid & 31;
  const int ln16 = lane & 15;
  const int hl   = lane >> 4;
  const int m0   = blockIdx.y * 128;
  const int n0   = blockIdx.x * 128;
  const int wm   = (wv >> 1) * 32;   // 0,32,64,96
  const int wn   = (wv & 1) * 64;    // 0,64

  // stage one 128x32 K-slab of A and Bt into LDS buffer `buf`
  auto stage = [&](int buf, int k0) {
#pragma unroll
    for (int i = 0; i < 2; ++i) {
      const int id = tid + i * 256;           // 0..511
      const int r  = id >> 2;                 // 0..127
      const int kc = (id & 3) << 3;           // 0,8,16,24
      const uint16_t* ga = &A [(size_t)(m0 + r) * K + k0 + kc];
      const uint16_t* gb = &Bt[(size_t)(n0 + r) * K + k0 + kc];
#if USE_ASYNC_LDS
      __builtin_amdgcn_global_load_async_to_lds_b128(
          (as1_i32x4*)const_cast<uint16_t*>(ga),
          (as3_i32x4*)&lsA[buf][r][kc], 0, 0);
      __builtin_amdgcn_global_load_async_to_lds_b128(
          (as1_i32x4*)const_cast<uint16_t*>(gb),
          (as3_i32x4*)&lsB[buf][r][kc], 0, 0);
#else
      *(uint4*)&lsA[buf][r][kc] = *(const uint4*)ga;
      *(uint4*)&lsB[buf][r][kc] = *(const uint4*)gb;
#endif
    }
  };

  f32x8 acc[2][4];
#pragma unroll
  for (int i = 0; i < 2; ++i)
#pragma unroll
    for (int j = 0; j < 4; ++j) acc[i][j] = zero8();

  stage(0, 0);
  int parity = 0;
  for (int k0 = 0; k0 < K; k0 += 32) {
    const bool hasNext = (k0 + 32 < K);
    if (hasNext) stage(parity ^ 1, k0 + 32);   // prefetch next slab into other buffer
#if USE_ASYNC_LDS
    // 4 async instrs per wave per stage; in-order retirement: <=4 outstanding
    // means the current buffer's loads have landed.
    if (hasNext) __builtin_amdgcn_s_wait_asynccnt(4);
    else         __builtin_amdgcn_s_wait_asynccnt(0);
#endif
    __syncthreads();                           // all waves' current-slab loads visible

    Frag fa[2], fb[4];
#pragma unroll
    for (int i = 0; i < 2; ++i) {
      const int r = wm + i * 16 + ln16;
      fa[i].u[0] = *(const uint4*)&lsA[parity][r][hl * 8];
      fa[i].u[1] = *(const uint4*)&lsA[parity][r][16 + hl * 8];
    }
#pragma unroll
    for (int j = 0; j < 4; ++j) {
      const int r = wn + j * 16 + ln16;
      fb[j].u[0] = *(const uint4*)&lsB[parity][r][hl * 8];
      fb[j].u[1] = *(const uint4*)&lsB[parity][r][16 + hl * 8];
    }
#pragma unroll
    for (int i = 0; i < 2; ++i)
#pragma unroll
      for (int j = 0; j < 4; ++j)
        acc[i][j] = __builtin_amdgcn_wmma_f32_16x16x32_bf16(
            false, fa[i].v, false, fb[j].v, (short)0, acc[i][j], false, false);
    __syncthreads();                           // readers done before buffer reuse
    parity ^= 1;
  }

  // epilogue
#pragma unroll
  for (int i = 0; i < 2; ++i) {
#pragma unroll
    for (int j = 0; j < 4; ++j) {
      const int gn = n0 + wn + j * 16 + ln16;
      const float bn = bias[gn];
#pragma unroll
      for (int r = 0; r < 8; ++r) {
        const int gm = m0 + wm + i * 16 + r + hl * 8;
        float c = acc[i][j][r] + bn;
        if constexpr (EPI == EPI_BIAS_GELU_BF16) c = gelu_new(c);
        if constexpr (EPI == EPI_BIAS_RES_F32) {
          ((float*)Out)[(size_t)gm * N + gn] = resid[(size_t)gm * N + gn] + c;
        } else {
          ((uint16_t*)Out)[(size_t)gm * N + gn] = f2bf(c);
        }
      }
    }
  }
}

// =====================================================================
// 4) causal flash attention, bf16 WMMA
//    grid: (B*H, T/64), block = 128 threads (4 waves), wave = 16 queries
//    qkv layout: [B*T][3C] bf16; attn out: [B*T][C] bf16
// =====================================================================
__global__ __launch_bounds__(128) void attention_kernel(
    const uint16_t* __restrict__ qkv, uint16_t* __restrict__ attn) {
  __shared__ uint16_t vt[64][40];        // V^T tile: [hd][key 0..31]
  __shared__ uint16_t pbuf[4][16][40];   // per-wave P: [q-row][key 0..31]
  const int bh = blockIdx.x;
  const int b  = bh >> 4;                // H = 16
  const int h  = bh & 15;
  const int q0 = blockIdx.y * 64;
  const int tid  = threadIdx.x;
  const int wv   = tid >> 5;
  const int lane = tid & 31;
  const int ln16 = lane & 15;
  const int hl   = lane >> 4;
  const int qw   = q0 + wv * 16;
  const size_t rowbase = (size_t)b * Tt * C3;

  // Q fragments (A layout), hd split into two K=32 slices, pre-loaded once
  Frag fq[2];
  {
    const size_t qrow = rowbase + (size_t)(qw + ln16) * C3 + h * 64;
#pragma unroll
    for (int f = 0; f < 2; ++f) {
      fq[f].u[0] = *(const uint4*)&qkv[qrow + f * 32 + hl * 8];
      fq[f].u[1] = *(const uint4*)&qkv[qrow + f * 32 + 16 + hl * 8];
    }
  }

  f32x8 o[4];
#pragma unroll
  for (int nt = 0; nt < 4; ++nt) o[nt] = zero8();
  float mrow[8], lrow[8];
#pragma unroll
  for (int r = 0; r < 8; ++r) { mrow[r] = -1e30f; lrow[r] = 0.0f; }
  const float scale = 0.125f;            // 1/sqrt(64)

  const int kend = q0 + 64;
  for (int kb = 0; kb < kend; kb += 32) {
    // ---- cooperative stage of V^T (32 keys x 64 hd) ----
#pragma unroll
    for (int i = 0; i < 2; ++i) {
      const int id = tid + i * 128;      // 0..255
      const int kr = id >> 3;            // key 0..31
      const int hc = (id & 7) * 8;       // hd chunk
      uint4 d = *(const uint4*)&qkv[rowbase + (size_t)(kb + kr) * C3 + 2 * Cc + h * 64 + hc];
      const uint16_t* e = (const uint16_t*)&d;
#pragma unroll
      for (int j = 0; j < 8; ++j) vt[hc + j][kr] = e[j];
    }

    // ---- scores for two 16-key tiles (Q @ K^T) ----
    f32x8 s[2];
#pragma unroll
    for (int j = 0; j < 2; ++j) {
      const int kk = kb + j * 16;
      const size_t krow = rowbase + (size_t)(kk + ln16) * C3 + Cc + h * 64;
      Frag fk[2];
#pragma unroll
      for (int f = 0; f < 2; ++f) {
        fk[f].u[0] = *(const uint4*)&qkv[krow + f * 32 + hl * 8];
        fk[f].u[1] = *(const uint4*)&qkv[krow + f * 32 + 16 + hl * 8];
      }
      f32x8 c = zero8();
      c = __builtin_amdgcn_wmma_f32_16x16x32_bf16(false, fq[0].v, false, fk[0].v, (short)0, c, false, false);
      c = __builtin_amdgcn_wmma_f32_16x16x32_bf16(false, fq[1].v, false, fk[1].v, (short)0, c, false, false);
      s[j] = c;
    }

    // ---- causal mask + online softmax update ----
#pragma unroll
    for (int r = 0; r < 8; ++r) {
      const int qi = qw + r + hl * 8;
      float a0 = (kb + ln16      <= qi) ? s[0][r] * scale : -1e30f;
      float a1 = (kb + 16 + ln16 <= qi) ? s[1][r] * scale : -1e30f;
      float mx = fmaxf(a0, a1);
#pragma unroll
      for (int ofs = 8; ofs; ofs >>= 1) mx = fmaxf(mx, __shfl_xor(mx, ofs, 16));
      const float mnew  = fmaxf(mrow[r], mx);
      const float alpha = __expf(mrow[r] - mnew);
      const float p0 = __expf(a0 - mnew);
      const float p1 = __expf(a1 - mnew);
      float sum = p0 + p1;
#pragma unroll
      for (int ofs = 8; ofs; ofs >>= 1) sum += __shfl_xor(sum, ofs, 16);
      lrow[r] = lrow[r] * alpha + sum;
      mrow[r] = mnew;
#pragma unroll
      for (int nt = 0; nt < 4; ++nt) o[nt][r] *= alpha;
      pbuf[wv][r + hl * 8][ln16]      = f2bf(p0);
      pbuf[wv][r + hl * 8][16 + ln16] = f2bf(p1);
    }
    __syncthreads();

    // ---- O += P @ V ----
    Frag fp;
    fp.u[0] = *(const uint4*)&pbuf[wv][ln16][hl * 8];
    fp.u[1] = *(const uint4*)&pbuf[wv][ln16][16 + hl * 8];
#pragma unroll
    for (int nt = 0; nt < 4; ++nt) {
      Frag fv;
      fv.u[0] = *(const uint4*)&vt[nt * 16 + ln16][hl * 8];
      fv.u[1] = *(const uint4*)&vt[nt * 16 + ln16][16 + hl * 8];
      o[nt] = __builtin_amdgcn_wmma_f32_16x16x32_bf16(
          false, fp.v, false, fv.v, (short)0, o[nt], false, false);
    }
    __syncthreads();
  }

  // ---- normalize and write out (bf16, [B*T][C] layout, head columns) ----
#pragma unroll
  for (int r = 0; r < 8; ++r) {
    const float inv = 1.0f / lrow[r];
    const int qi = qw + r + hl * 8;
    const size_t obase = ((size_t)b * Tt + qi) * Cc + h * 64;
#pragma unroll
    for (int nt = 0; nt < 4; ++nt)
      attn[obase + nt * 16 + ln16] = f2bf(o[nt][r] * inv);
  }
}

// =====================================================================
// host-side launch
// =====================================================================
extern "C" void kernel_launch(void* const* d_in, const int* in_sizes, int n_in,
                              void* d_out, int out_size, void* d_ws, size_t ws_size,
                              hipStream_t stream) {
  (void)in_sizes; (void)n_in; (void)out_size; (void)ws_size;
  const float* x      = (const float*)d_in[0];
  const float* ln1_g  = (const float*)d_in[1];
  const float* ln1_b  = (const float*)d_in[2];
  const float* w_qkv  = (const float*)d_in[3];
  const float* b_qkv  = (const float*)d_in[4];
  const float* w_o    = (const float*)d_in[5];
  const float* b_o    = (const float*)d_in[6];
  const float* ln2_g  = (const float*)d_in[7];
  const float* ln2_b  = (const float*)d_in[8];
  const float* w_fc   = (const float*)d_in[9];
  const float* b_fc   = (const float*)d_in[10];
  const float* w_proj = (const float*)d_in[11];
  const float* b_proj = (const float*)d_in[12];

  char* ws = (char*)d_ws;
  uint16_t* wqkvT = (uint16_t*)(ws + OFF_WQKVT);
  uint16_t* woT   = (uint16_t*)(ws + OFF_WOT);
  uint16_t* wfcT  = (uint16_t*)(ws + OFF_WFCT);
  uint16_t* wprjT = (uint16_t*)(ws + OFF_WPRJT);
  uint16_t* hbuf  = (uint16_t*)(ws + OFF_H);
  uint16_t* qkvb  = (uint16_t*)(ws + OFF_QKV);
  uint16_t* gbuf  = (uint16_t*)(ws + OFF_QKV);   // reuse (qkv dead after attention)
  uint16_t* attnb = (uint16_t*)(ws + OFF_ATTN);
  float*    x1    = (float*)   (ws + OFF_X1);

  // weight prep: W[K][N] fp32 -> Wt[N][K] bf16
  transpose_cvt_kernel<<<dim3(C3 / 32,   Cc / 32),   256, 0, stream>>>(w_qkv,  wqkvT, Cc,   C3);
  transpose_cvt_kernel<<<dim3(Cc / 32,   Cc / 32),   256, 0, stream>>>(w_o,    woT,   Cc,   Cc);
  transpose_cvt_kernel<<<dim3(DFFd / 32, Cc / 32),   256, 0, stream>>>(w_fc,   wfcT,  Cc,   DFFd);
  transpose_cvt_kernel<<<dim3(Cc / 32,   DFFd / 32), 256, 0, stream>>>(w_proj, wprjT, DFFd, Cc);

  // h = LN1(x) -> bf16
  layernorm_kernel<<<Mrows, 256, 0, stream>>>(x, ln1_g, ln1_b, hbuf);

  // qkv = h @ w_qkv + b_qkv  -> bf16 [M][3C]
  gemm_bf16_kernel<EPI_BIAS_BF16><<<dim3(C3 / 128, Mrows / 128), 256, 0, stream>>>(
      hbuf, wqkvT, b_qkv, nullptr, qkvb, Mrows, C3, Cc);

  // causal flash attention -> bf16 [M][C]
  attention_kernel<<<dim3(Bb * Hh, Tt / 64), 128, 0, stream>>>(qkvb, attnb);

  // x1 = x + attn @ w_o + b_o  -> fp32
  gemm_bf16_kernel<EPI_BIAS_RES_F32><<<dim3(Cc / 128, Mrows / 128), 256, 0, stream>>>(
      attnb, woT, b_o, x, x1, Mrows, Cc, Cc);

  // h2 = LN2(x1) -> bf16 (reuse hbuf)
  layernorm_kernel<<<Mrows, 256, 0, stream>>>(x1, ln2_g, ln2_b, hbuf);

  // g = gelu(h2 @ w_fc + b_fc) -> bf16 [M][DFF]
  gemm_bf16_kernel<EPI_BIAS_GELU_BF16><<<dim3(DFFd / 128, Mrows / 128), 256, 0, stream>>>(
      hbuf, wfcT, b_fc, nullptr, gbuf, Mrows, DFFd, Cc);

  // out = x1 + g @ w_proj + b_proj -> fp32
  gemm_bf16_kernel<EPI_BIAS_RES_F32><<<dim3(Cc / 128, Mrows / 128), 256, 0, stream>>>(
      gbuf, wprjT, b_proj, x1, (float*)d_out, Mrows, Cc, DFFd);
}
